// MyLSTMCell_85512798863714
// MI455X (gfx1250) — compile-verified, matrix operand order checked by reading, not run on hardware
//
#include <hip/hip_runtime.h>
#include <hip/hip_bf16.h>
#include <math.h>

// ---------------------------------------------------------------------------
// Layernorm-LSTM cell for MI455X (gfx1250, wave32, WMMA).
//   z = x@Wx + bx + h@Wh + bh            (bf16 WMMA, f32 accumulate)
//   f = sig(LN0(zf)); i = sig(LN1(f));   (source bug replicated)
//   g = tanh(LN2(zg)); o = sig(LN3(zo));
//   h_new = o * tanh(c*f + i*g)
// Compute-bound (~1200 FLOP/B) -> v_wmma_f32_16x16x32_bf16, f32 accumulate.
// Round 3: K-loop split into x-phase / h-phase (no per-iter operand select);
// per-thread base pointers advanced by constant strides so all global loads
// use immediate IOFFSETs (24-bit) — minimal per-iteration address VALU.
// Double-buffered LDS ping-pong, one barrier per K-step.
// Workspace: z buffer 4096*8192*4 = 128 MB (fits in 192 MB L2).
// ---------------------------------------------------------------------------

#define BATCH   4096
#define IN_DIM  2048
#define OUT_DIM 2048
#define NCOLS   (4 * OUT_DIM)   // 8192
#define BM 128
#define BN 128
#define BK 32
#define KSTEPS_PER_PHASE (IN_DIM / BK)   // 64
#define LPITCH 48               // bf16 elements; 96 B row pitch (16B aligned)
#define ATILE (BM * LPITCH)
#define BTILE (BN * LPITCH)

typedef __attribute__((ext_vector_type(16))) __bf16 v16bf;
typedef __attribute__((ext_vector_type(8)))  __bf16 v8bf;
typedef __attribute__((ext_vector_type(4)))  __bf16 v4bf;
typedef __attribute__((ext_vector_type(8)))  float  v8f;

// ---- pipeline stages -------------------------------------------------------
// All loads at constant offsets from one per-thread base pointer.
__device__ __forceinline__ void prefetch_a(const float* __restrict__ ap,
                                           float4 (&a)[4]) {
#pragma unroll
    for (int it = 0; it < 4; ++it)
        a[it] = *reinterpret_cast<const float4*>(ap + (size_t)it * 32 * IN_DIM);
}

__device__ __forceinline__ void prefetch_b(const float* __restrict__ bp,
                                           float (&b)[4][4]) {
#pragma unroll
    for (int it = 0; it < 4; ++it)
#pragma unroll
        for (int j = 0; j < 4; ++j)
            b[it][j] = bp[(size_t)(8 * it + j) * NCOLS];
}

__device__ __forceinline__ void stage_store(__bf16* __restrict__ Abuf,
                                            __bf16* __restrict__ Bbuf,
                                            const float4 (&a)[4],
                                            const float (&b)[4][4],
                                            int ar0, int akq, int bn0, int bk0) {
#pragma unroll
    for (int it = 0; it < 4; ++it) {
        const int r = ar0 + (it << 5);
        v4bf p;
        p[0] = (__bf16)a[it].x; p[1] = (__bf16)a[it].y;
        p[2] = (__bf16)a[it].z; p[3] = (__bf16)a[it].w;
        *reinterpret_cast<v4bf*>(&Abuf[r * LPITCH + (akq << 2)]) = p;
    }
#pragma unroll
    for (int it = 0; it < 4; ++it) {
        const int kq = bk0 + (it << 1);
        v4bf p;
#pragma unroll
        for (int j = 0; j < 4; ++j) p[j] = (__bf16)b[it][j];
        *reinterpret_cast<v4bf*>(&Bbuf[bn0 * LPITCH + (kq << 2)]) = p;
    }
}

__device__ __forceinline__ void compute_tile(const __bf16* __restrict__ Abuf,
                                             const __bf16* __restrict__ Bbuf,
                                             v8f (&acc)[2][4],
                                             int wm, int wn, int l15, int half) {
    // lanes 0-15 hold K 0..7 & 16..23, lanes 16-31 hold K 8..15 & 24..31
    v16bf afr[2];
#pragma unroll
    for (int i = 0; i < 2; ++i) {
        const __bf16* base = &Abuf[(wm * 32 + i * 16 + l15) * LPITCH + (half << 3)];
        v8bf lo = *reinterpret_cast<const v8bf*>(base);
        v8bf hi = *reinterpret_cast<const v8bf*>(base + 16);
        afr[i] = __builtin_shufflevector(lo, hi, 0, 1, 2, 3, 4, 5, 6, 7,
                                         8, 9, 10, 11, 12, 13, 14, 15);
    }
    v16bf bfr[4];
#pragma unroll
    for (int j = 0; j < 4; ++j) {
        const __bf16* base = &Bbuf[(wn * 64 + j * 16 + l15) * LPITCH + (half << 3)];
        v8bf lo = *reinterpret_cast<const v8bf*>(base);
        v8bf hi = *reinterpret_cast<const v8bf*>(base + 16);
        bfr[j] = __builtin_shufflevector(lo, hi, 0, 1, 2, 3, 4, 5, 6, 7,
                                         8, 9, 10, 11, 12, 13, 14, 15);
    }
#pragma unroll
    for (int i = 0; i < 2; ++i)
#pragma unroll
        for (int j = 0; j < 4; ++j)
            acc[i][j] = __builtin_amdgcn_wmma_f32_16x16x32_bf16(
                false, afr[i], false, bfr[j], (short)0, acc[i][j], false, false);
}

// ----------------------------- GEMM kernel ---------------------------------
__global__ __launch_bounds__(256)
void lstm_gemm_bf16(const float* __restrict__ x,
                    const float* __restrict__ h,
                    const float* __restrict__ Wx,
                    const float* __restrict__ Wh,
                    const float* __restrict__ bx,
                    const float* __restrict__ bh,
                    float* __restrict__ z) {
    __shared__ __bf16 Asm[2 * ATILE];   // [buf][row][k], k contiguous
    __shared__ __bf16 Bsm[2 * BTILE];   // [buf][col][k], k contiguous (B^T)

    const int tid   = threadIdx.x;
    const int mBase = blockIdx.y * BM;
    const int nBase = blockIdx.x * BN;

    // wave layout: 8 waves as 4 (M) x 2 (N); each wave owns 32x64
    const int wave = tid >> 5;
    const int lane = tid & 31;
    const int wm   = wave >> 1;          // 0..3
    const int wn   = wave & 1;           // 0..1
    const int half = lane >> 4;          // 0/1
    const int l15  = lane & 15;

    const int ar0 = tid >> 3;            // A loader: row 0..31 (+32*it)
    const int akq = tid & 7;             // A loader: k-quad 0..7
    const int bn0 = tid & 127;           // B loader: col 0..127
    const int bk0 = tid >> 7;            // B loader: k-quad base 0..1

    v8f acc[2][4];
#pragma unroll
    for (int i = 0; i < 2; ++i)
#pragma unroll
        for (int j = 0; j < 4; ++j)
            acc[i][j] = (v8f){0.f, 0.f, 0.f, 0.f, 0.f, 0.f, 0.f, 0.f};

    float4 aReg[4];
    float  bReg[4][4];

    // per-thread base pointers (advance by constant stride per K-step)
    const size_t aOff = (size_t)(mBase + ar0) * IN_DIM + (akq << 2);
    const size_t bOff = (size_t)(bk0 << 2) * NCOLS + nBase + bn0;

    // ---- prologue: stage x-phase K-step 0 into buffer 0 ----
    const float* ap = x + aOff;
    const float* bp = Wx + bOff;
    prefetch_a(ap, aReg);
    prefetch_b(bp, bReg);
    stage_store(Asm, Bsm, aReg, bReg, ar0, akq, bn0, bk0);
    __syncthreads();

    int cur = 0;
    // ---- phase 1: x @ Wx, K-steps 1..63 ----
    for (int s = 1; s < KSTEPS_PER_PHASE; ++s) {
        ap += BK;
        bp += (size_t)BK * NCOLS;
        prefetch_a(ap, aReg);
        prefetch_b(bp, bReg);
        compute_tile(Asm + cur * ATILE, Bsm + cur * BTILE, acc, wm, wn, l15, half);
        stage_store(Asm + (cur ^ 1) * ATILE, Bsm + (cur ^ 1) * BTILE,
                    aReg, bReg, ar0, akq, bn0, bk0);
        __syncthreads();
        cur ^= 1;
    }

    // ---- boundary: prefetch h-phase K-step 0 while computing last x tile ----
    ap = h + aOff;
    bp = Wh + bOff;
    prefetch_a(ap, aReg);
    prefetch_b(bp, bReg);
    compute_tile(Asm + cur * ATILE, Bsm + cur * BTILE, acc, wm, wn, l15, half);
    stage_store(Asm + (cur ^ 1) * ATILE, Bsm + (cur ^ 1) * BTILE,
                aReg, bReg, ar0, akq, bn0, bk0);
    __syncthreads();
    cur ^= 1;

    // ---- phase 2: h @ Wh, K-steps 1..63 ----
    for (int s = 1; s < KSTEPS_PER_PHASE; ++s) {
        ap += BK;
        bp += (size_t)BK * NCOLS;
        prefetch_a(ap, aReg);
        prefetch_b(bp, bReg);
        compute_tile(Asm + cur * ATILE, Bsm + cur * BTILE, acc, wm, wn, l15, half);
        stage_store(Asm + (cur ^ 1) * ATILE, Bsm + (cur ^ 1) * BTILE,
                    aReg, bReg, ar0, akq, bn0, bk0);
        __syncthreads();
        cur ^= 1;
    }
    compute_tile(Asm + cur * ATILE, Bsm + cur * BTILE, acc, wm, wn, l15, half);

    // ---- epilogue: add bias, write z ----
#pragma unroll
    for (int i = 0; i < 2; ++i)
#pragma unroll
        for (int j = 0; j < 4; ++j) {
            const int col  = nBase + wn * 64 + j * 16 + l15;
            const float bias = bx[col] + bh[col];
            const int rowb = mBase + wm * 32 + i * 16 + half * 8;
#pragma unroll
            for (int e = 0; e < 8; ++e)
                z[(size_t)(rowb + e) * NCOLS + col] = acc[i][j][e] + bias;
        }
}

// --------------------------- epilogue kernel --------------------------------
__device__ __forceinline__ void block_reduce2(float& s1, float& s2) {
    __shared__ float buf1[8], buf2[8];
#pragma unroll
    for (int off = 16; off > 0; off >>= 1) {
        s1 += __shfl_xor(s1, off, 32);
        s2 += __shfl_xor(s2, off, 32);
    }
    const int lane = threadIdx.x & 31;
    const int w    = threadIdx.x >> 5;
    if (lane == 0) { buf1[w] = s1; buf2[w] = s2; }
    __syncthreads();
    float t1 = 0.f, t2 = 0.f;
#pragma unroll
    for (int k = 0; k < 8; ++k) { t1 += buf1[k]; t2 += buf2[k]; }
    __syncthreads();           // protect buffers for the next call
    s1 = t1; s2 = t2;
}

__device__ __forceinline__ void layernorm8(float (&v)[8],
                                           const float* __restrict__ gamma,
                                           const float* __restrict__ beta,
                                           int tid) {
    float s = 0.f, s2 = 0.f;
#pragma unroll
    for (int e = 0; e < 8; ++e) { s += v[e]; s2 += v[e] * v[e]; }
    block_reduce2(s, s2);
    const float mu  = s * (1.f / (float)OUT_DIM);
    const float var = s2 * (1.f / (float)OUT_DIM) - mu * mu;
    const float rs  = rsqrtf(var + 1e-5f);
#pragma unroll
    for (int e = 0; e < 8; ++e) {
        const int col = e * 256 + tid;
        v[e] = (v[e] - mu) * rs * gamma[col] + beta[col];
    }
}

__device__ __forceinline__ float sigmoidf_(float x) {
    return 1.f / (1.f + __expf(-x));
}

__global__ __launch_bounds__(256)
void lstm_epilogue(const float* __restrict__ z,
                   const float* __restrict__ c,
                   const float* __restrict__ gamma,
                   const float* __restrict__ beta,
                   float* __restrict__ hout) {
    const int row = blockIdx.x;
    const int tid = threadIdx.x;
    const float* zrow = z + (size_t)row * NCOLS;

    float f[8], iv[8], g[8], o[8];

    // f-gate
#pragma unroll
    for (int e = 0; e < 8; ++e) f[e] = zrow[0 * OUT_DIM + e * 256 + tid];
    layernorm8(f, gamma + 0 * OUT_DIM, beta + 0 * OUT_DIM, tid);
#pragma unroll
    for (int e = 0; e < 8; ++e) f[e] = sigmoidf_(f[e]);

    // i-gate (source bug: layernorm of activated f, not z_i)
#pragma unroll
    for (int e = 0; e < 8; ++e) iv[e] = f[e];
    layernorm8(iv, gamma + 1 * OUT_DIM, beta + 1 * OUT_DIM, tid);
#pragma unroll
    for (int e = 0; e < 8; ++e) iv[e] = sigmoidf_(iv[e]);

    // g-gate
#pragma unroll
    for (int e = 0; e < 8; ++e) g[e] = zrow[2 * OUT_DIM + e * 256 + tid];
    layernorm8(g, gamma + 2 * OUT_DIM, beta + 2 * OUT_DIM, tid);
#pragma unroll
    for (int e = 0; e < 8; ++e) g[e] = tanhf(g[e]);

    // o-gate
#pragma unroll
    for (int e = 0; e < 8; ++e) o[e] = zrow[3 * OUT_DIM + e * 256 + tid];
    layernorm8(o, gamma + 3 * OUT_DIM, beta + 3 * OUT_DIM, tid);
#pragma unroll
    for (int e = 0; e < 8; ++e) o[e] = sigmoidf_(o[e]);

    // combine
#pragma unroll
    for (int e = 0; e < 8; ++e) {
        const int col = e * 256 + tid;
        const float cn = c[(size_t)row * OUT_DIM + col] * f[e] + iv[e] * g[e];
        hout[(size_t)row * OUT_DIM + col] = o[e] * tanhf(cn);
    }
}

// ------------------------------- launcher -----------------------------------
extern "C" void kernel_launch(void* const* d_in, const int* in_sizes, int n_in,
                              void* d_out, int out_size, void* d_ws, size_t ws_size,
                              hipStream_t stream) {
    const float* x     = (const float*)d_in[0];
    const float* h     = (const float*)d_in[1];
    const float* c     = (const float*)d_in[2];
    const float* Wx    = (const float*)d_in[3];
    const float* bx    = (const float*)d_in[4];
    const float* Wh    = (const float*)d_in[5];
    const float* bh    = (const float*)d_in[6];
    const float* gamma = (const float*)d_in[7];
    const float* beta  = (const float*)d_in[8];
    float* hout = (float*)d_out;
    float* z    = (float*)d_ws;   // needs BATCH*NCOLS*4 = 128 MB

    dim3 gridG(NCOLS / BN, BATCH / BM);   // (64, 32)
    lstm_gemm_bf16<<<gridG, 256, 0, stream>>>(x, h, Wx, Wh, bx, bh, z);
    lstm_epilogue<<<BATCH, 256, 0, stream>>>(z, c, gamma, beta, hout);
}
